// RouteMoELayer_56839597195652
// MI455X (gfx1250) — matrix operand order repaired and checked by reading, not compiled, not run on hardware
//
#include <hip/hip_runtime.h>
#include <math.h>

typedef __attribute__((ext_vector_type(16))) __bf16 v16bf;
typedef __attribute__((ext_vector_type(8)))  float  v8f;
typedef __attribute__((ext_vector_type(4)))  float  v4f;
typedef unsigned int u32x4 __attribute__((ext_vector_type(4)));
typedef int          i32x4 __attribute__((ext_vector_type(4)));
typedef int          i32x8 __attribute__((ext_vector_type(8)));

#define B_DIM 128
#define T_DIM 32
#define H_DIM 768
#define E_DIM 4
#define F_DIM 3072
#define NB    2
#define NROWS (NB*B_DIM)              /* 256 beam rows */
#define FC    64                      /* F-chunk width */
#define NCHUNK (F_DIM/FC)             /* 48 */

#define OUT_MAIN (NROWS*T_DIM*H_DIM)  /* 6291456 floats of 'output' */
/* then: beam_scores[256], expert_route[256], beam_idx[256], importance_loss[1] */

#define CHUNK_ELEMS (FC*H_DIM)        /* 49152 bf16 = 98304 bytes per staged chunk */
#define WT_ELEMS ((size_t)E_DIM*F_DIM*H_DIM)   /* per converted weight tensor */

// ---------------------------------------------------------------- helpers
static __device__ __forceinline__ __bf16 f2bf(float f) {
    union { float f; unsigned u; } x; x.f = f;
    unsigned r = (x.u + 0x7FFFu + ((x.u >> 16) & 1u)) >> 16;  // RNE
    union { unsigned short s; __bf16 b; } y; y.s = (unsigned short)r;
    return y.b;
}

// A-matrix fragment (16x32 bf16), base is row-major [m][stride]
static __device__ __forceinline__ v16bf frag_a(const __bf16* base, int stride,
                                               int mt, int kc, int lane) {
    int m  = mt * 16 + (lane & 15);
    int k0 = kc * 32 + ((lane >> 4) << 3);          // 0 or 8
    const __bf16* p = base + m * stride + k0;
    union { v16bf v; v4f f[2]; } u;
    u.f[0] = *(const v4f*)(p);                      // K k0..k0+7
    u.f[1] = *(const v4f*)(p + 16);                 // K k0+16..k0+23
    return u.v;
}

// B-matrix fragment (32x16 bf16), base is [n][stride] (K contiguous per column n)
static __device__ __forceinline__ v16bf frag_b(const __bf16* base, int stride,
                                               int nt, int kc, int lane) {
    int n  = nt * 16 + (lane & 15);
    int k0 = kc * 32 + ((lane >> 4) << 3);          // 0 or 8
    const __bf16* p = base + n * stride + k0;
    union { v16bf v; v4f f[2]; } u;
    u.f[0] = *(const v4f*)(p);
    u.f[1] = *(const v4f*)(p + 16);
    return u.v;
}

// TDM: one contiguous 98304-byte block, global -> LDS (1-D tile of 12288 x 8B)
static __device__ __forceinline__ void tdm_load_chunk(const __bf16* gsrc, __bf16* ldst) {
    unsigned long long ga = (unsigned long long)(uintptr_t)gsrc;
    unsigned lds = (unsigned)(uintptr_t)ldst;
    u32x4 g0;
    g0.x = 1u;                                   // count=1 (valid user descriptor)
    g0.y = lds;                                  // lds_addr (bytes)
    g0.z = (unsigned)ga;                         // global_addr[31:0]
    g0.w = (unsigned)(ga >> 32) | 0x80000000u;   // global_addr[56:32] | type=2
    i32x8 g1;
    g1[0] = 0x00030000;                          // data_size=3 (8B); mask/flags = 0
    g1[1] = (int)(12288u << 16);                 // tensor_dim0[15:0] in [31:16]
    g1[2] = (int)(1u << 16);                     // tensor_dim0 hi=0 | tensor_dim1=1
    g1[3] = (int)(12288u << 16);                 // tensor_dim1 hi=0 | tile_dim0=12288
    g1[4] = 1;                                   // tile_dim1=1, tile_dim2=0
    g1[5] = 12288;                               // tensor_dim0_stride[31:0]
    g1[6] = 0;                                   // stride hi / dim1_stride lo
    g1[7] = 0;
    i32x4 z4 = {0, 0, 0, 0};
    i32x8 z8 = {0, 0, 0, 0, 0, 0, 0, 0};
    __builtin_amdgcn_tensor_load_to_lds(g0, g1, z4, z4, z8, 0);
}

// ---------------------------------------------------------------- weight prep
// W1 (E,H,F) f32  ->  W1t (E,F,H) bf16   (each F-chunk contiguous, B-frag order)
__global__ __launch_bounds__(256)
void prep_w1(const float* __restrict__ W1, __bf16* __restrict__ W1t) {
    __shared__ float tile[32][33];
    int e = blockIdx.z;
    int f0 = blockIdx.x * 32, h0 = blockIdx.y * 32;
    int tx = threadIdx.x, ty = threadIdx.y;          // 32 x 8
    const float* src = W1 + ((size_t)e * H_DIM + h0) * F_DIM + f0;
    #pragma unroll
    for (int j = 0; j < 4; ++j)
        tile[ty + 8*j][tx] = src[(size_t)(ty + 8*j) * F_DIM + tx]; // tile[h][f]
    __syncthreads();
    __bf16* dst = W1t + ((size_t)e * F_DIM + f0) * H_DIM + h0;
    #pragma unroll
    for (int j = 0; j < 4; ++j)
        dst[(size_t)(ty + 8*j) * H_DIM + tx] = f2bf(tile[tx][ty + 8*j]);
}

// W2 (E,F,H) f32 -> W2t (E, fc, H, f_local) bf16  (each F-chunk contiguous)
__global__ __launch_bounds__(256)
void prep_w2(const float* __restrict__ W2, __bf16* __restrict__ W2t) {
    __shared__ float tile[32][33];
    int e = blockIdx.z;
    int h0 = blockIdx.x * 32, f0 = blockIdx.y * 32;
    int tx = threadIdx.x, ty = threadIdx.y;
    const float* src = W2 + ((size_t)e * F_DIM + f0) * H_DIM + h0;
    #pragma unroll
    for (int j = 0; j < 4; ++j)
        tile[ty + 8*j][tx] = src[(size_t)(ty + 8*j) * H_DIM + tx]; // tile[f][h]
    __syncthreads();
    int f  = f0 + tx;
    int fcb = f >> 6, fl = f & (FC - 1);
    #pragma unroll
    for (int j = 0; j < 4; ++j) {
        int h = h0 + ty + 8*j;
        W2t[(((size_t)e * NCHUNK + fcb) * H_DIM + h) * FC + fl] = f2bf(tile[tx][ty + 8*j]);
    }
}

// ---------------------------------------------------------------- gating
__global__ __launch_bounds__(256)
void gate_kernel(const float* __restrict__ x, const float* __restrict__ Wg,
                 float* __restrict__ out, float* __restrict__ ws_prob,
                 int* __restrict__ ws_route, float* __restrict__ ws_score) {
    int b   = blockIdx.x;
    int tid = threadIdx.x;
    __shared__ float red[E_DIM][256];
    float acc[E_DIM] = {0.f, 0.f, 0.f, 0.f};
    const float* xb = x + (size_t)b * T_DIM * H_DIM;
    for (int h = tid; h < H_DIM; h += 256) {
        float s = 0.f;
        #pragma unroll 8
        for (int t = 0; t < T_DIM; ++t) s += xb[t * H_DIM + h];
        s *= (1.0f / (float)T_DIM);
        #pragma unroll
        for (int e = 0; e < E_DIM; ++e) acc[e] += s * Wg[e * H_DIM + h];
    }
    #pragma unroll
    for (int e = 0; e < E_DIM; ++e) red[e][tid] = acc[e];
    __syncthreads();
    for (int off = 128; off > 0; off >>= 1) {
        if (tid < off) {
            #pragma unroll
            for (int e = 0; e < E_DIM; ++e) red[e][tid] += red[e][tid + off];
        }
        __syncthreads();
    }
    if (tid == 0) {
        float p[E_DIM];
        float mx = -1e30f;
        #pragma unroll
        for (int e = 0; e < E_DIM; ++e) mx = fmaxf(mx, red[e][0]);
        float sum = 0.f;
        #pragma unroll
        for (int e = 0; e < E_DIM; ++e) { p[e] = __expf(red[e][0] - mx); sum += p[e]; }
        float inv = 1.0f / sum;
        #pragma unroll
        for (int e = 0; e < E_DIM; ++e) { p[e] *= inv; ws_prob[b * E_DIM + e] = p[e]; }
        int i0 = 0;
        for (int e = 1; e < E_DIM; ++e) if (p[e] > p[i0]) i0 = e;
        int i1 = -1;
        for (int e = 0; e < E_DIM; ++e)
            if (e != i0 && (i1 < 0 || p[e] > p[i1])) i1 = e;
        int r0 = 2 * b, r1 = 2 * b + 1;
        out[OUT_MAIN + r0]       = p[i0];
        out[OUT_MAIN + r1]       = p[i1];
        out[OUT_MAIN + 256 + r0] = (float)i0;
        out[OUT_MAIN + 256 + r1] = (float)i1;
        out[OUT_MAIN + 512 + r0] = (float)r0;
        out[OUT_MAIN + 512 + r1] = (float)r1;
        ws_route[r0] = i0;   ws_route[r1] = i1;
        ws_score[r0] = p[i0]; ws_score[r1] = p[i1];
    }
}

// ---------------------------------------------------------------- importance loss
__global__ void loss_kernel(const float* __restrict__ ws_prob, float* __restrict__ out) {
    if (threadIdx.x == 0 && blockIdx.x == 0) {
        float imp[E_DIM] = {0.f, 0.f, 0.f, 0.f};
        for (int b = 0; b < B_DIM; ++b)
            #pragma unroll
            for (int e = 0; e < E_DIM; ++e) imp[e] += ws_prob[b * E_DIM + e];
        float mean = 0.25f * (imp[0] + imp[1] + imp[2] + imp[3]);
        float var = 0.f;
        #pragma unroll
        for (int e = 0; e < E_DIM; ++e) { float d = imp[e] - mean; var += d * d; }
        var *= (1.0f / (float)(E_DIM - 1));          // ddof=1
        out[OUT_MAIN + 768] = var / (mean * mean);
    }
}

// ---------------------------------------------------------------- expert FFN
__global__ __launch_bounds__(256)
void ffn_kernel(const float* __restrict__ x,
                const __bf16* __restrict__ W1t, const float* __restrict__ b1,
                const __bf16* __restrict__ W2t, const float* __restrict__ b2,
                const int* __restrict__ route, const float* __restrict__ scorep,
                float* __restrict__ out) {
    __shared__ __bf16 Xs[T_DIM * H_DIM];   // 48 KB : X row-major [m][k]
    __shared__ __bf16 Wbuf[CHUNK_ELEMS];   // 96 KB : TDM target (W1 then W2 chunk)
    __shared__ __bf16 H1s[T_DIM * FC];     //  4 KB : gelu'd H1 chunk [m][f]

    int r    = blockIdx.x;
    int tid  = threadIdx.x;
    int lane = tid & 31;
    int w    = tid >> 5;
    int e    = route[r];
    float score = scorep[r];
    int borig = r >> 1;

    const float*  xb   = x   + (size_t)borig * T_DIM * H_DIM;
    const __bf16* W1te = W1t + (size_t)e * F_DIM * H_DIM;   // [f][h], chunk-contig
    const __bf16* W2te = W2t + (size_t)e * F_DIM * H_DIM;   // [fc][h][fl], chunk-contig
    const float*  b1e  = b1  + (size_t)e * F_DIM;
    const float*  b2e  = b2  + (size_t)e * H_DIM;

    // stage X as bf16 (vectorized: b128 read -> 4x cvt -> b64 LDS write)
    {
        const v4f* xv = (const v4f*)xb;
        for (int i = tid; i < T_DIM * H_DIM / 4; i += 256) {
            v4f v = xv[i];
            union { __bf16 h[4]; unsigned long long u; } pk;
            pk.h[0] = f2bf(v[0]); pk.h[1] = f2bf(v[1]);
            pk.h[2] = f2bf(v[2]); pk.h[3] = f2bf(v[3]);
            ((unsigned long long*)Xs)[i] = pk.u;
        }
    }

    v8f acc[6][2];
    #pragma unroll
    for (int j = 0; j < 6; ++j) { acc[j][0] = (v8f)(0.f); acc[j][1] = (v8f)(0.f); }

    int mt1 = w >> 2, nt1 = w & 3;

    for (int fcc = 0; fcc < NCHUNK; ++fcc) {
        __syncthreads();                       // prior consumers of Wbuf/H1s done
        if (w == 0) {                          // wave 0 drives the TDM
            tdm_load_chunk(W1te + (size_t)fcc * CHUNK_ELEMS, Wbuf);
            __builtin_amdgcn_s_wait_tensorcnt(0);
        }
        __syncthreads();
        // ---- GEMM1: each wave one 16x16 tile of H1 chunk
        {
            v8f c = (v8f)(0.f);
            for (int kc = 0; kc < H_DIM / 32; ++kc) {
                v16bf a  = frag_a(Xs,   H_DIM, mt1, kc, lane);
                v16bf bm = frag_b(Wbuf, H_DIM, nt1, kc, lane);
                c = __builtin_amdgcn_wmma_f32_16x16x32_bf16(
                        false, a, false, bm, (short)0, c, false, false);
            }
            int nl   = nt1 * 16 + (lane & 15);
            float bb = b1e[fcc * FC + nl];
            int mrow = mt1 * 16 + ((lane >> 4) << 3);
            #pragma unroll
            for (int i = 0; i < 8; ++i) {
                float v = c[i] + bb;
                v = 0.5f * v * (1.0f + erff(v * 0.70710678118654752f));
                H1s[(mrow + i) * FC + nl] = f2bf(v);
            }
        }
        __syncthreads();
        if (w == 0) {
            tdm_load_chunk(W2te + (size_t)fcc * CHUNK_ELEMS, Wbuf);
            __builtin_amdgcn_s_wait_tensorcnt(0);
        }
        __syncthreads();
        // ---- GEMM2 accumulate: OUT += H1s(32xFC) @ W2chunk(FCx768)
        #pragma unroll
        for (int kc = 0; kc < FC / 32; ++kc) {
            v16bf a0 = frag_a(H1s, FC, 0, kc, lane);
            v16bf a1 = frag_a(H1s, FC, 1, kc, lane);
            #pragma unroll
            for (int j = 0; j < 6; ++j) {
                int nt = w + 8 * j;
                v16bf bm = frag_b(Wbuf, FC, nt, kc, lane);
                acc[j][0] = __builtin_amdgcn_wmma_f32_16x16x32_bf16(
                                false, a0, false, bm, (short)0, acc[j][0], false, false);
                acc[j][1] = __builtin_amdgcn_wmma_f32_16x16x32_bf16(
                                false, a1, false, bm, (short)0, acc[j][1], false, false);
            }
        }
    }

    // ---- epilogue: + b2, * score, store f32
    float* outr = out + (size_t)r * T_DIM * H_DIM;
    int nl   = lane & 15;
    int moff = (lane >> 4) << 3;
    #pragma unroll
    for (int j = 0; j < 6; ++j) {
        int n    = (w + 8 * j) * 16 + nl;
        float bb = b2e[n];
        #pragma unroll
        for (int mt = 0; mt < 2; ++mt) {
            #pragma unroll
            for (int i = 0; i < 8; ++i) {
                int m = mt * 16 + moff + i;
                outr[m * H_DIM + n] = (acc[j][mt][i] + bb) * score;
            }
        }
    }
}

// ---------------------------------------------------------------- launcher
extern "C" void kernel_launch(void* const* d_in, const int* in_sizes, int n_in,
                              void* d_out, int out_size, void* d_ws, size_t ws_size,
                              hipStream_t stream) {
    const float* x  = (const float*)d_in[0];
    const float* Wg = (const float*)d_in[1];
    const float* W1 = (const float*)d_in[2];
    const float* b1 = (const float*)d_in[3];
    const float* W2 = (const float*)d_in[4];
    const float* b2 = (const float*)d_in[5];
    float* out = (float*)d_out;

    // workspace: [W1t bf16 | W2t bf16 | gating scratch]  (~37.75 MB + 4 KB)
    __bf16* W1t = (__bf16*)d_ws;
    __bf16* W2t = (__bf16*)d_ws + WT_ELEMS;
    char*  gb   = (char*)d_ws + 2 * WT_ELEMS * sizeof(__bf16);
    float* ws_prob  = (float*)gb;                 // 512 floats
    int*   ws_route = (int*)(gb + 512 * 4);       // 256 ints
    float* ws_score = (float*)(gb + 768 * 4);     // 256 floats

    prep_w1<<<dim3(F_DIM/32, H_DIM/32, E_DIM), dim3(32, 8), 0, stream>>>(W1, W1t);
    prep_w2<<<dim3(H_DIM/32, F_DIM/32, E_DIM), dim3(32, 8), 0, stream>>>(W2, W2t);
    gate_kernel<<<B_DIM, 256, 0, stream>>>(x, Wg, out, ws_prob, ws_route, ws_score);
    loss_kernel<<<1, 32, 0, stream>>>(ws_prob, out);
    ffn_kernel<<<NROWS, 256, 0, stream>>>(x, W1t, b1, W2t, b2, ws_route, ws_score, out);
}
